// DynamicPointerDecoder_85306640433386
// MI455X (gfx1250) — matrix operand order repaired and checked by reading, not compiled, not run on hardware
//
#include <hip/hip_runtime.h>
#include <cstdint>
#include <cstddef>

// ---------------- problem constants ----------------
#define HD     200
#define POOL   16
#define TLEN   4096
#define NITER  4
#define NP     (POOL * HD)   // 3200 rows of W1/W2 (p*200+h)
#define K1P    416           // 400 (U part of layer1 K) padded to mult of 32
#define K2P    224           // 200 padded to mult of 32

typedef __attribute__((ext_vector_type(16))) __bf16 v16bf;
typedef __attribute__((ext_vector_type(8)))  float  v8f;
typedef unsigned int u32x4 __attribute__((ext_vector_type(4)));
typedef int          i32x8 __attribute__((ext_vector_type(8)));
typedef int          i32x4 __attribute__((ext_vector_type(4)));
typedef __bf16 bf16;

__device__ __forceinline__ float sigm(float x) { return 1.0f / (1.0f + expf(-x)); }

// ---- TDM: DMA a (rows x cols) bf16 tile (row stride = strideElems) into LDS ----
// D# per CDNA5 ISA 8.3/8.4: group0 = {count/type/lds_addr/global_addr},
// group1 = {data_size=2B, tensor dims, tile dims, dim0 stride}; 2-D -> groups 2/3 zero.
// This toolchain's builtin takes 6 args (g0, g1, g2, g3, g4, cpol).
__device__ __forceinline__ void tdm_load_tile_b16(const bf16* gsrc, unsigned ldsOff,
                                                  int rows, int cols, int strideElems) {
  unsigned long long ga = (unsigned long long)(uintptr_t)gsrc;
  u32x4 g0;
  g0[0] = 1u;                                             // count=1, user mode, no gather
  g0[1] = ldsOff;                                         // lds_addr (bytes)
  g0[2] = (unsigned)ga;                                   // global_addr[31:0]
  g0[3] = (unsigned)((ga >> 32) & 0x01FFFFFFu) | (2u << 30); // addr[56:32] | type=2
  i32x8 g1;
  g1[0] = 1 << 16;                                        // data_size=1 (2 bytes), mask=0
  g1[1] = (int)(((unsigned)strideElems & 0xFFFFu) << 16); // tensor_dim0[15:0]
  g1[2] = (int)(((unsigned)strideElems >> 16) | ((unsigned)rows << 16)); // d0 hi | dim1 lo
  g1[3] = (int)((unsigned)cols << 16);                    // dim1 hi=0 | tile_dim0
  g1[4] = rows;                                           // tile_dim1 (tile_dim2=0)
  g1[5] = strideElems;                                    // tensor_dim0_stride[31:0]
  g1[6] = 0;
  g1[7] = 0;
  i32x4 z4; z4[0] = 0; z4[1] = 0; z4[2] = 0; z4[3] = 0;
  i32x8 z8; z8[0] = 0; z8[1] = 0; z8[2] = 0; z8[3] = 0;
  z8[4] = 0; z8[5] = 0; z8[6] = 0; z8[7] = 0;
  __builtin_amdgcn_tensor_load_to_lds(g0, g1, z4, z4, z8, 0);
}

// ---------------- conversion / padding kernels ----------------
__global__ __launch_bounds__(256) void cvt_U_kernel(const float* __restrict__ U,
                                                    bf16* __restrict__ Ubf) {
  int idx = blockIdx.x * 256 + threadIdx.x;           // over TLEN*K1P (exact)
  int t = idx / K1P, k = idx % K1P;
  float v = (k < 400) ? U[(size_t)k * TLEN + t] : 0.0f;
  Ubf[idx] = (bf16)v;
}

__global__ __launch_bounds__(256) void cvt_pad_kernel(const float* __restrict__ src,
                                                      bf16* __restrict__ dst,
                                                      int rows, int srcStride, int Kv, int Kp) {
  int idx = blockIdx.x * 256 + threadIdx.x;
  if (idx >= rows * Kp) return;
  int r = idx / Kp, k = idx % Kp;
  float v = (k < Kv) ? src[(size_t)r * srcStride + k] : 0.0f;
  dst[idx] = (bf16)v;
}

// ---------------- WMMA bf16 GEMM:  C(MxN) = A(MxKp) * B(NxKp)^T ----------------
// 128x128 block tile, 8 waves (2x4), wave tile 64x32 (4x2 wmma), K step 32.
// LDS tiles are double-buffered and filled by the Tensor Data Mover (wave 0 issues,
// s_wait_tensorcnt + barrier publishes), overlapping DMA of tile k+1 with WMMA of tile k.
template <bool OUT_BF16>
__global__ __launch_bounds__(256) void gemm_bf16_kernel(const bf16* __restrict__ A,
                                                        const bf16* __restrict__ B,
                                                        void* __restrict__ C,
                                                        int N, int Kp) {
  __shared__ bf16 As[2][128][32];
  __shared__ bf16 Bs[2][128][32];

  const int tid  = threadIdx.x;
  const int lane = tid & 31;
  const int wave = tid >> 5;        // 0..7
  const int wm   = wave >> 2;       // 0..1  (64-row strip)
  const int wn   = wave & 3;        // 0..3  (32-col strip)
  const int bm   = blockIdx.x * 128;
  const int bn   = blockIdx.y * 128;
  const int lrow = lane & 15;
  const int lhi  = lane >> 4;

  v8f acc[4][2];
  const v8f vzero = {0.f, 0.f, 0.f, 0.f, 0.f, 0.f, 0.f, 0.f};
  for (int i = 0; i < 4; ++i)
    for (int j = 0; j < 2; ++j) acc[i][j] = vzero;

  const bf16* Ablk = A + (size_t)bm * Kp;
  const bf16* Bblk = B + (size_t)bn * Kp;
  const int nk = Kp / 32;

  if (wave == 0) {
    tdm_load_tile_b16(Ablk, (unsigned)(uintptr_t)&As[0][0][0], 128, 32, Kp);
    tdm_load_tile_b16(Bblk, (unsigned)(uintptr_t)&Bs[0][0][0], 128, 32, Kp);
  }

  for (int ki = 0; ki < nk; ++ki) {
    const int cur = ki & 1;
    if ((wave == 0) && (ki + 1 < nk)) {
      const int nxt = cur ^ 1;
      tdm_load_tile_b16(Ablk + (ki + 1) * 32, (unsigned)(uintptr_t)&As[nxt][0][0], 128, 32, Kp);
      tdm_load_tile_b16(Bblk + (ki + 1) * 32, (unsigned)(uintptr_t)&Bs[nxt][0][0], 128, 32, Kp);
      __builtin_amdgcn_s_wait_tensorcnt(2);   // tile ki landed (TENSORcnt is in-order)
    } else {
      __builtin_amdgcn_s_wait_tensorcnt(0);   // non-issuer waves: cnt==0, no-op
    }
    __syncthreads();                          // publish LDS tile to all waves

    // B fragments (contiguous order used here: element e -> K = lhi*16 + e)
    v16bf bfrag[2];
    for (int j = 0; j < 2; ++j) {
      const bf16* pb = &Bs[cur][wn * 32 + j * 16 + lrow][lhi * 16];
      ((uint4*)&bfrag[j])[0] = *(const uint4*)pb;
      ((uint4*)&bfrag[j])[1] = *(const uint4*)(pb + 8);
    }
    for (int i = 0; i < 4; ++i) {
      // A layout: element e -> K = lhi*16 + e (matches B)
      const int am = wm * 64 + i * 16 + lrow;
      v16bf afrag;
      ((uint4*)&afrag)[0] = *(const uint4*)&As[cur][am][lhi * 16];
      ((uint4*)&afrag)[1] = *(const uint4*)&As[cur][am][lhi * 16 + 8];
      for (int j = 0; j < 2; ++j)
        acc[i][j] = __builtin_amdgcn_wmma_f32_16x16x32_bf16(
            false, afrag, false, bfrag[j], (short)0, acc[i][j], false, false);
    }
    __syncthreads();                          // done reading before buffer is re-filled
  }

  // write back: vgpr r -> row lhi*8+r, col lrow
  for (int i = 0; i < 4; ++i)
    for (int j = 0; j < 2; ++j) {
      const int gm0 = bm + wm * 64 + i * 16 + lhi * 8;
      const int gn  = bn + wn * 32 + j * 16 + lrow;
      for (int r = 0; r < 8; ++r) {
        float v = acc[i][j][r];
        if (OUT_BF16)
          ((bf16*)C)[(size_t)(gm0 + r) * N + gn] = (bf16)v;
        else
          ((float*)C)[(size_t)(gm0 + r) * N + gn] = v;
      }
    }
}

// ---------------- serial part: LSTM step + r_a, r_b (one workgroup) ----------------
__global__ __launch_bounds__(256) void lstm_kernel(
    const float* __restrict__ U, const float* __restrict__ h0, const float* __restrict__ c0,
    const float* __restrict__ W_ih, const float* __restrict__ W_hh,
    const float* __restrict__ b_ih, const float* __restrict__ b_hh,
    const float* __restrict__ WDa, const float* __restrict__ WDb,
    float* __restrict__ hS, float* __restrict__ cS,
    float* __restrict__ rA, float* __restrict__ rB,
    const int* __restrict__ sIdx, const int* __restrict__ eIdx, int iter) {
  __shared__ float x[800];          // [u_s(400), u_e(400)]
  __shared__ float hP[HD], cP[HD], g[800], hN[HD];
  const int tid = threadIdx.x;
  const int s = (iter == 0) ? 0 : *sIdx;
  const int e = (iter == 0) ? 0 : *eIdx;
  for (int k = tid; k < 400; k += 256) {
    x[k]       = U[(size_t)k * TLEN + s];
    x[400 + k] = U[(size_t)k * TLEN + e];
  }
  for (int k = tid; k < HD; k += 256) {
    hP[k] = (iter == 0) ? h0[k] : hS[k];
    cP[k] = (iter == 0) ? c0[k] : cS[k];
  }
  __syncthreads();
  for (int j = tid; j < 800; j += 256) {
    float a = b_ih[j] + b_hh[j];
    const float* wi = W_ih + (size_t)j * 800;
    for (int k = 0; k < 800; ++k) a += wi[k] * x[k];
    const float* wh = W_hh + (size_t)j * HD;
    for (int k = 0; k < HD; ++k) a += wh[k] * hP[k];
    g[j] = a;
  }
  __syncthreads();
  for (int j = tid; j < HD; j += 256) {
    float ig = sigm(g[j]), fg = sigm(g[200 + j]);
    float gg = tanhf(g[400 + j]), og = sigm(g[600 + j]);
    float cv = fg * cP[j] + ig * gg;
    float hv = og * tanhf(cv);
    cS[j] = cv; hS[j] = hv; hN[j] = hv;
  }
  __syncthreads();
  for (int j = tid; j < HD; j += 256) {
    float aa = 0.f, bb = 0.f;
    const float* wa = WDa + (size_t)j * 1000;
    const float* wb = WDb + (size_t)j * 1000;
    for (int k = 0; k < HD; ++k) { aa += wa[k] * hN[k]; bb += wb[k] * hN[k]; }
    for (int k = 0; k < 800; ++k) { aa += wa[200 + k] * x[k]; bb += wb[200 + k] * x[k]; }
    rA[j] = tanhf(aa); rB[j] = tanhf(bb);
  }
}

// biasEff[j] = b1[j] + sum_d r[d] * W1[j, 400+d]   (folds the broadcast-r columns)
__global__ __launch_bounds__(256) void bias_kernel(const float* __restrict__ W1,
                                                   const float* __restrict__ b1,
                                                   const float* __restrict__ r,
                                                   float* __restrict__ biasEff) {
  int j = blockIdx.x * 256 + threadIdx.x;
  if (j >= NP) return;
  float a = b1[j];
  const float* w = W1 + (size_t)j * 600 + 400;
  for (int k = 0; k < HD; ++k) a += w[k] * r[k];
  biasEff[j] = a;
}

// m1[t,h] = max_p (G1[t, p*200+h] + biasEff[p*200+h]); also bf16 padded copy for GEMM2
__global__ __launch_bounds__(256) void max1_kernel(const bf16* __restrict__ G1,
                                                   const float* __restrict__ biasEff,
                                                   float* __restrict__ m1,
                                                   bf16* __restrict__ m1b) {
  int idx = blockIdx.x * 256 + threadIdx.x;   // over TLEN*K2P (exact)
  int t = idx / K2P, h = idx % K2P;
  if (h >= HD) { m1b[idx] = (bf16)0.0f; return; }
  const bf16* row = G1 + (size_t)t * NP;
  float best = -3.0e38f;
  for (int p = 0; p < POOL; ++p) {
    float v = (float)row[p * HD + h] + biasEff[p * HD + h];
    best = fmaxf(best, v);
  }
  m1[(size_t)t * HD + h] = best;
  m1b[idx] = (bf16)best;
}

// m2[t,h] = max_p (C2[t, p*200+h] + b2[p*200+h])
__global__ __launch_bounds__(256) void max2_kernel(const float* __restrict__ C2,
                                                   const float* __restrict__ b2,
                                                   float* __restrict__ m2) {
  int idx = blockIdx.x * 256 + threadIdx.x;   // over TLEN*HD (exact)
  int t = idx / HD, h = idx % HD;
  const float* row = C2 + (size_t)t * NP;
  float best = -3.0e38f;
  for (int p = 0; p < POOL; ++p) best = fmaxf(best, row[p * HD + h] + b2[p * HD + h]);
  m2[idx] = best;
}

// scores[t] = max_p ( m12[t] . W3[p] + b3[p] )
__global__ __launch_bounds__(256) void score_kernel(const float* __restrict__ m1,
                                                    const float* __restrict__ m2,
                                                    const float* __restrict__ W3,
                                                    const float* __restrict__ b3,
                                                    float* __restrict__ out) {
  int t = blockIdx.x * 256 + threadIdx.x;     // TLEN threads (exact)
  const float* a = m1 + (size_t)t * HD;
  const float* b = m2 + (size_t)t * HD;
  float best = -3.0e38f;
  for (int p = 0; p < POOL; ++p) {
    const float* w = W3 + (size_t)p * 400;
    float s = b3[p];
    for (int d = 0; d < HD; ++d) s += a[d] * w[d];
    for (int d = 0; d < HD; ++d) s += b[d] * w[200 + d];
    best = fmaxf(best, s);
  }
  out[t] = best;
}

__global__ __launch_bounds__(256) void argmax_kernel(const float* __restrict__ scores,
                                                     int n, int* __restrict__ outIdx) {
  __shared__ float sv[256];
  __shared__ int   si[256];
  int tid = threadIdx.x;
  float bv = -3.0e38f; int bi = 0;
  for (int i = tid; i < n; i += 256) {
    float v = scores[i];
    if (v > bv) { bv = v; bi = i; }           // strided indices increase -> first hit wins
  }
  sv[tid] = bv; si[tid] = bi;
  __syncthreads();
  for (int st = 128; st > 0; st >>= 1) {
    if (tid < st) {
      if (sv[tid + st] > sv[tid] || (sv[tid + st] == sv[tid] && si[tid + st] < si[tid])) {
        sv[tid] = sv[tid + st]; si[tid] = si[tid + st];
      }
    }
    __syncthreads();
  }
  if (tid == 0) *outIdx = si[0];
}

__global__ void write_se_kernel(const int* __restrict__ s, const int* __restrict__ e,
                                float* __restrict__ out) {
  if (threadIdx.x == 0) { out[0] = (float)(*s); out[1] = (float)(*e); }
}

// ---------------- host orchestration ----------------
static inline char* wsAlloc(char*& p, size_t bytes) {
  char* r = p;
  p += (bytes + 255) & ~(size_t)255;
  return r;
}

extern "C" void kernel_launch(void* const* d_in, const int* in_sizes, int n_in,
                              void* d_out, int out_size, void* d_ws, size_t ws_size,
                              hipStream_t stream) {
  (void)in_sizes; (void)n_in; (void)out_size; (void)ws_size;
  const float* U    = (const float*)d_in[0];
  const float* h0   = (const float*)d_in[2];
  const float* c0   = (const float*)d_in[3];
  const float* W_ih = (const float*)d_in[4];
  const float* W_hh = (const float*)d_in[5];
  const float* b_ih = (const float*)d_in[6];
  const float* b_hh = (const float*)d_in[7];
  const float* WD[2] = {(const float*)d_in[8],  (const float*)d_in[15]};
  const float* W1[2] = {(const float*)d_in[9],  (const float*)d_in[16]};
  const float* b1[2] = {(const float*)d_in[10], (const float*)d_in[17]};
  const float* W2[2] = {(const float*)d_in[11], (const float*)d_in[18]};
  const float* b2[2] = {(const float*)d_in[12], (const float*)d_in[19]};
  const float* W3[2] = {(const float*)d_in[13], (const float*)d_in[20]};
  const float* b3[2] = {(const float*)d_in[14], (const float*)d_in[21]};
  float* out = (float*)d_out;

  char* p = (char*)d_ws;
  bf16*  Ubf     = (bf16*)wsAlloc(p, (size_t)TLEN * K1P * sizeof(bf16));
  bf16*  W1p[2]  = {(bf16*)wsAlloc(p, (size_t)NP * K1P * sizeof(bf16)),
                    (bf16*)wsAlloc(p, (size_t)NP * K1P * sizeof(bf16))};
  bf16*  W2p[2]  = {(bf16*)wsAlloc(p, (size_t)NP * K2P * sizeof(bf16)),
                    (bf16*)wsAlloc(p, (size_t)NP * K2P * sizeof(bf16))};
  bf16*  G1[2]   = {(bf16*)wsAlloc(p, (size_t)TLEN * NP * sizeof(bf16)),
                    (bf16*)wsAlloc(p, (size_t)TLEN * NP * sizeof(bf16))};
  float* C2      = (float*)wsAlloc(p, (size_t)TLEN * NP * sizeof(float));
  float* m1f[2]  = {(float*)wsAlloc(p, (size_t)TLEN * HD * sizeof(float)),
                    (float*)wsAlloc(p, (size_t)TLEN * HD * sizeof(float))};
  bf16*  m1b[2]  = {(bf16*)wsAlloc(p, (size_t)TLEN * K2P * sizeof(bf16)),
                    (bf16*)wsAlloc(p, (size_t)TLEN * K2P * sizeof(bf16))};
  float* m2f[2]  = {(float*)wsAlloc(p, (size_t)TLEN * HD * sizeof(float)),
                    (float*)wsAlloc(p, (size_t)TLEN * HD * sizeof(float))};
  float* hS      = (float*)wsAlloc(p, HD * sizeof(float));
  float* cS      = (float*)wsAlloc(p, HD * sizeof(float));
  float* rv[2]   = {(float*)wsAlloc(p, HD * sizeof(float)),
                    (float*)wsAlloc(p, HD * sizeof(float))};
  float* bEff[2] = {(float*)wsAlloc(p, NP * sizeof(float)),
                    (float*)wsAlloc(p, NP * sizeof(float))};
  int*   sIdx    = (int*)wsAlloc(p, sizeof(int));
  int*   eIdx    = (int*)wsAlloc(p, sizeof(int));

  const dim3 gGrid(TLEN / 128, NP / 128);   // (32, 25)

  // ---- one-time precompute: bf16 operands + G1 = Ubf * W1[:, :400]^T (per head)
  cvt_U_kernel<<<(TLEN * K1P) / 256, 256, 0, stream>>>(U, Ubf);
  for (int h = 0; h < 2; ++h) {
    cvt_pad_kernel<<<(NP * K1P) / 256, 256, 0, stream>>>(W1[h], W1p[h], NP, 600, 400, K1P);
    cvt_pad_kernel<<<(NP * K2P) / 256, 256, 0, stream>>>(W2[h], W2p[h], NP, 200, 200, K2P);
  }
  for (int h = 0; h < 2; ++h)
    gemm_bf16_kernel<true><<<gGrid, 256, 0, stream>>>(Ubf, W1p[h], G1[h], NP, K1P);

  // ---- 4 sequential decoder iterations
  for (int it = 0; it < NITER; ++it) {
    lstm_kernel<<<1, 256, 0, stream>>>(U, h0, c0, W_ih, W_hh, b_ih, b_hh,
                                       WD[0], WD[1], hS, cS, rv[0], rv[1], sIdx, eIdx, it);
    for (int h = 0; h < 2; ++h)
      bias_kernel<<<(NP + 255) / 256, 256, 0, stream>>>(W1[h], b1[h], rv[h], bEff[h]);

    for (int h = 0; h < 2; ++h) {
      max1_kernel<<<(TLEN * K2P) / 256, 256, 0, stream>>>(G1[h], bEff[h], m1f[h], m1b[h]);
      gemm_bf16_kernel<false><<<gGrid, 256, 0, stream>>>(m1b[h], W2p[h], C2, NP, K2P);
      max2_kernel<<<(TLEN * HD) / 256, 256, 0, stream>>>(C2, b2[h], m2f[h]);
      float* row = out + (size_t)h * NITER * TLEN + (size_t)it * TLEN;
      score_kernel<<<TLEN / 256, 256, 0, stream>>>(m1f[h], m2f[h], W3[h], b3[h], row);
      argmax_kernel<<<1, 256, 0, stream>>>(row, TLEN, h == 0 ? sIdx : eIdx);
    }
  }
  write_se_kernel<<<1, 32, 0, stream>>>(sIdx, eIdx, out + 2 * NITER * TLEN);
}
